// CausalTransformerEncoderLayer_10299331575847
// MI455X (gfx1250) — matrix operand order repaired and checked by reading, not compiled, not run on hardware
//
#include <hip/hip_runtime.h>

// ---------------------------------------------------------------------------
// Causal transformer encoder layer for MI455X (gfx1250).
// B=4, S=2048, D=512, F=2048, H=8, hd=64.
// bf16 WMMA GEMMs (f32 accumulate) with TDM-staged weight tiles in LDS,
// flash attention (no score materialization), fused bias/ReLU/LN epilogues.
// ---------------------------------------------------------------------------

#define D_MODEL 512
#define NHEADS  8
#define HDIM    64
#define SEQ     2048
#define BATCH   4
#define FDIM    2048
#define NTOK    (BATCH * SEQ)      // 8192 tokens
#define QKV_LD  (3 * D_MODEL)      // 1536

typedef __attribute__((ext_vector_type(16))) __bf16          v16bf;
typedef __attribute__((ext_vector_type(8)))  float           v8f;
typedef __attribute__((ext_vector_type(8)))  unsigned short  u16x8;
typedef __attribute__((ext_vector_type(16))) unsigned short  u16x16;

union U16V {
  u16x16 v;
  struct { u16x8 lo, hi; };
  v16bf  b;
};

__device__ __forceinline__ unsigned short f2bf(float f) {
  unsigned u = __float_as_uint(f);
  u += 0x7FFFu + ((u >> 16) & 1u);           // round-to-nearest-even
  return (unsigned short)(u >> 16);
}

// A-matrix 16x32 bf16 fragment (ISA 7.12.2): lane<16 -> row m=lane, K {0..7,16..23};
// lane>=16 -> row m=lane-16, K {8..15,24..31}.
__device__ __forceinline__ v16bf load_a_frag(const unsigned short* row0, int ldk, int lane) {
  const int m  = lane & 15;
  const int hi = lane >> 4;
  const unsigned short* p = row0 + (size_t)m * ldk + (hi ? 8 : 0);
  U16V r;
  r.lo = *(const u16x8*)(p);
  r.hi = *(const u16x8*)(p + 16);
  return r.b;
}

// B-matrix 32x16 bf16 fragment: lane<16 -> column n=lane, K {0..7,16..23}; lane>=16 ->
// column n=lane-16, K {8..15,24..31}. Two 16B loads so only 16B alignment is required.
__device__ __forceinline__ v16bf load_b_frag(const unsigned short* col0, int ldk, int lane) {
  const int n  = lane & 15;
  const int ko = (lane >> 4) << 3;
  const unsigned short* p = col0 + (size_t)n * ldk + ko;
  U16V r;
  r.lo = *(const u16x8*)(p);
  r.hi = *(const u16x8*)(p + 16);
  return r.b;
}

#define WMMA_BF16(A, B, C) \
  __builtin_amdgcn_wmma_f32_16x16x32_bf16(false, (A), false, (B), (short)0, (C), false, false)

// ---------------------------------------------------------------------------
// f32 -> bf16 cast (grid-stride)
// ---------------------------------------------------------------------------
__global__ void cast_bf16_kernel(const float* __restrict__ s, unsigned short* __restrict__ d, int n) {
  for (int i = blockIdx.x * blockDim.x + threadIdx.x; i < n; i += gridDim.x * blockDim.x)
    d[i] = f2bf(s[i]);
}

// ---------------------------------------------------------------------------
// GEMM: Out[M,N] = act(A[M,K](bf16) @ W[N,K](bf16)^T + bias[N])
// Workgroup = 8 waves, block tile 256x64; each wave owns a 32x64 tile
// (8 wmma per K=32 step). The 64xK weight slab is shared by all waves and
// staged into double-buffered LDS in 128-element K-chunks by the Tensor Data
// Mover (TENSORcnt / s_wait_tensorcnt), with D#.pad so LDS rows stride 272B
// (bank-conflict friendly). Falls back to direct global reads if the TDM
// builtin is unavailable.
// ---------------------------------------------------------------------------
#define GEMM_MB    256
#define GEMM_NB    64
#define GEMM_CHUNK 128                       // K elements per staged chunk
#define GEMM_LROW  (GEMM_CHUNK + 8)          // +16B pad per row -> 272B stride

#if __has_builtin(__builtin_amdgcn_tensor_load_to_lds)
#define USE_TDM 1
#else
#define USE_TDM 0
#endif

#if USE_TDM
typedef __attribute__((ext_vector_type(4))) unsigned int u32x4;
typedef __attribute__((ext_vector_type(8))) int          i32x8;
typedef __attribute__((ext_vector_type(4))) int          i32x4;

// Issue a TDM load of a [64 rows x GEMM_CHUNK elems] bf16 tile (row stride =
// ldk elements in memory) into LDS at lds_byte (row stride GEMM_LROW elems).
// D# per CDNA5 ISA ch.8: group0 = {count,lds,global,type}, group1 = dims/strides.
__device__ __forceinline__ void tdm_load_b_tile(unsigned lds_byte,
                                                const unsigned short* gsrc, int ldk) {
  const unsigned long long ga = (unsigned long long)(size_t)gsrc;
  u32x4 g0;
  g0[0] = 1u;                                          // count=1 (valid user desc)
  g0[1] = lds_byte;                                    // lds_addr (bytes)
  g0[2] = (unsigned)ga;                                // global_addr[31:0]
  g0[3] = (unsigned)(ga >> 32) | (2u << 30);           // global_addr[56:32] | type=2
  i32x8 g1;
  // data_size=2B (code 1), pad_enable=1, pad_interval=32 qwords (code 5 = one
  // 256B row), pad_amount=4 dwords (code 3) -> LDS row stride 256+16 bytes.
  g1[0] = (int)((1u << 16) | (1u << 20) | (5u << 22) | (3u << 25));
  g1[1] = (int)(((unsigned)ldk & 0xFFFFu) << 16);      // tensor_dim0[15:0]
  g1[2] = (int)(64u << 16);                            // tensor_dim1 = 64 rows
  g1[3] = (int)((unsigned)GEMM_CHUNK << 16);           // tile_dim0 = 128 elems
  g1[4] = 64;                                          // tile_dim1 = 64 rows
  g1[5] = ldk;                                         // tensor_dim0_stride
  g1[6] = 0;
  g1[7] = 0;
  i32x4 g2 = {1, 0, 0, 0};                             // tensor_dim2=1 (benign)
  i32x4 g3 = {0, 0, 0, 0};
#if __has_include(<hip/amd_detail/amd_gfx1250_TDM.h>)
  i32x8 g4 = {0, 0, 0, 0, 0, 0, 0, 0};
  __builtin_amdgcn_tensor_load_to_lds(g0, g1, g2, g3, g4, 0);
#else
  __builtin_amdgcn_tensor_load_to_lds(g0, g1, g2, g3, 0);
#endif
}
#endif  // USE_TDM

template <bool RELU, bool WRITE_F32, bool WRITE_BF16>
__global__ __launch_bounds__(256) void gemm_wmma_kernel(
    const unsigned short* __restrict__ A, const unsigned short* __restrict__ W,
    const float* __restrict__ bias, float* __restrict__ Cf,
    unsigned short* __restrict__ Cb, int M, int N, int K) {
  const int lane = threadIdx.x & 31;
  const int wv   = threadIdx.x >> 5;
  const int nbn  = N / GEMM_NB;
  const int tn   = (blockIdx.x % nbn) * GEMM_NB;
  const int tm   = (blockIdx.x / nbn) * GEMM_MB + wv * 32;

  v8f acc[2][4] = {};
  const unsigned short* a0 = A + (size_t)tm * K;
  const unsigned short* a1 = a0 + (size_t)16 * K;
  const unsigned short* w0 = W + (size_t)tn * K;

#if USE_TDM
  __shared__ unsigned short bstage[2][GEMM_NB * GEMM_LROW];
  const int nch = K / GEMM_CHUNK;
  if (wv == 0)
    tdm_load_b_tile((unsigned)(size_t)(const void*)&bstage[0][0], w0, K);
  for (int c = 0; c < nch; ++c) {
    if (wv == 0) {
      if (c + 1 < nch) {
        tdm_load_b_tile((unsigned)(size_t)(const void*)&bstage[(c + 1) & 1][0],
                        w0 + (c + 1) * GEMM_CHUNK, K);
        __builtin_amdgcn_s_wait_tensorcnt(1);   // chunk c landed (in-order)
      } else {
        __builtin_amdgcn_s_wait_tensorcnt(0);
      }
    }
    __syncthreads();
    const unsigned short* bl = &bstage[c & 1][0];
    const int kc = c * GEMM_CHUNK;
    if (c + 1 < nch) {
      __builtin_prefetch(a0 + kc + GEMM_CHUNK, 0, 1);  // global_prefetch_b8
      __builtin_prefetch(a1 + kc + GEMM_CHUNK, 0, 1);
    }
#pragma unroll
    for (int lk = 0; lk < GEMM_CHUNK; lk += 32) {
      v16bf fa0 = load_a_frag(a0 + kc + lk, K, lane);
      v16bf fa1 = load_a_frag(a1 + kc + lk, K, lane);
      v16bf fb0 = load_b_frag(bl + lk,                 GEMM_LROW, lane);
      v16bf fb1 = load_b_frag(bl + 16 * GEMM_LROW + lk, GEMM_LROW, lane);
      v16bf fb2 = load_b_frag(bl + 32 * GEMM_LROW + lk, GEMM_LROW, lane);
      v16bf fb3 = load_b_frag(bl + 48 * GEMM_LROW + lk, GEMM_LROW, lane);
      acc[0][0] = WMMA_BF16(fa0, fb0, acc[0][0]);
      acc[0][1] = WMMA_BF16(fa0, fb1, acc[0][1]);
      acc[0][2] = WMMA_BF16(fa0, fb2, acc[0][2]);
      acc[0][3] = WMMA_BF16(fa0, fb3, acc[0][3]);
      acc[1][0] = WMMA_BF16(fa1, fb0, acc[1][0]);
      acc[1][1] = WMMA_BF16(fa1, fb1, acc[1][1]);
      acc[1][2] = WMMA_BF16(fa1, fb2, acc[1][2]);
      acc[1][3] = WMMA_BF16(fa1, fb3, acc[1][3]);
    }
    __syncthreads();
  }
#else
  for (int k = 0; k < K; k += 32) {
    v16bf fa0 = load_a_frag(a0 + k, K, lane);
    v16bf fa1 = load_a_frag(a1 + k, K, lane);
#pragma unroll
    for (int t = 0; t < 4; ++t) {
      v16bf fb = load_b_frag(w0 + (size_t)(t * 16) * K + k, K, lane);
      acc[0][t] = WMMA_BF16(fa0, fb, acc[0][t]);
      acc[1][t] = WMMA_BF16(fa1, fb, acc[1][t]);
    }
  }
#endif

  const int nloc = lane & 15;
  const int hoff = (lane >> 4) << 3;         // C/D: upper half-wave holds rows M=8..15
  float bi[4];
#pragma unroll
  for (int t = 0; t < 4; ++t) bi[t] = bias[tn + t * 16 + nloc];
#pragma unroll
  for (int i = 0; i < 2; ++i) {
#pragma unroll
    for (int r = 0; r < 8; ++r) {
      const size_t m = (size_t)tm + i * 16 + hoff + r;
#pragma unroll
      for (int t = 0; t < 4; ++t) {
        float v = acc[i][t][r] + bi[t];
        if (RELU) v = fmaxf(v, 0.f);
        const size_t o = m * N + tn + t * 16 + nloc;
        if (WRITE_F32)  Cf[o] = v;
        if (WRITE_BF16) Cb[o] = f2bf(v);
      }
    }
  }
}

// ---------------------------------------------------------------------------
// V transpose: vt[b][h][d][s] = qkv_bf16[(b*S+s)*1536 + 1024 + h*64 + d]
// ---------------------------------------------------------------------------
__global__ void vtrans_kernel(const unsigned short* __restrict__ qkvb,
                              unsigned short* __restrict__ vt) {
  const int idx = blockIdx.x * blockDim.x + threadIdx.x; // BATCH*NHEADS*HDIM*SEQ
  const int s = idx & (SEQ - 1);
  const int d = (idx >> 11) & (HDIM - 1);
  const int h = (idx >> 17) & (NHEADS - 1);
  const int b = idx >> 20;
  vt[idx] = qkvb[((size_t)(b * SEQ + s)) * QKV_LD + 2 * D_MODEL + h * HDIM + d];
}

// ---------------------------------------------------------------------------
// Flash attention: one wave per (b, h, 16-query-row tile). Streams 32-wide KV
// tiles with online softmax; P relaid via per-wave LDS slab (stride 40 to
// break bank conflicts) into an A-fragment for the P@V WMMAs.
// ---------------------------------------------------------------------------
#define ATT_WAVES 4
__global__ __launch_bounds__(128) void attention_kernel(
    const unsigned short* __restrict__ qkvb, const unsigned short* __restrict__ vt,
    unsigned short* __restrict__ ctxb) {
  __shared__ unsigned short pshare[ATT_WAVES][16 * 40];

  const int lane = threadIdx.x & 31;
  const int wv   = threadIdx.x >> 5;
  const int gw   = blockIdx.x * ATT_WAVES + wv;     // 0..4095
  const int qt   = gw & 127;                        // SEQ/16 query tiles
  const int bh   = gw >> 7;                         // 0..31
  const int b = bh >> 3, h = bh & 7;
  const int qbase = qt * 16;
  const int tok0  = b * SEQ;

  const unsigned short* qrow = qkvb + (size_t)(tok0 + qbase) * QKV_LD + h * HDIM;
  const unsigned short* krow = qkvb + (size_t)tok0 * QKV_LD + D_MODEL + h * HDIM;
  const unsigned short* vrow = vt + (size_t)bh * HDIM * SEQ;
  unsigned short* pl = pshare[wv];

  const v16bf q0 = load_a_frag(qrow,      QKV_LD, lane);   // head-dim K 0..31
  const v16bf q1 = load_a_frag(qrow + 32, QKV_LD, lane);   // head-dim K 32..63

  v8f c0 = {}, c1 = {}, c2 = {}, c3 = {};                  // ctx [16 x 64]
  float mrow[8], lrow[8];
#pragma unroll
  for (int r = 0; r < 8; ++r) { mrow[r] = -1e30f; lrow[r] = 0.f; }

  const float scale = 0.125f;                              // 1/sqrt(64)
  const int nloc = lane & 15;
  const int hoff = (lane >> 4) << 3;
  const int jmax = (qbase + 16 + 31) >> 5;                 // 32-wide kv tiles

  for (int j = 0; j < jmax; ++j) {
    const int kv0 = j * 32;
    if (j + 1 < jmax) {                                    // prefetch next KV tile
      __builtin_prefetch(krow + (size_t)(kv0 + 32 + nloc) * QKV_LD, 0, 1);
      __builtin_prefetch(vrow + (size_t)nloc * 4 * SEQ + kv0 + 32, 0, 1);
    }
    // ---- scores S[16 x 32] = Q @ K^T : K-dim = 64 head dims, two N tiles ----
    v8f sa = {}, sb = {};
    {
      const unsigned short* kc0 = krow + (size_t)kv0 * QKV_LD;        // kv cols 0..15
      const unsigned short* kc1 = krow + (size_t)(kv0 + 16) * QKV_LD; // kv cols 16..31
      sa = WMMA_BF16(q0, load_b_frag(kc0,      QKV_LD, lane), sa);
      sa = WMMA_BF16(q1, load_b_frag(kc0 + 32, QKV_LD, lane), sa);
      sb = WMMA_BF16(q0, load_b_frag(kc1,      QKV_LD, lane), sb);
      sb = WMMA_BF16(q1, load_b_frag(kc1 + 32, QKV_LD, lane), sb);
    }
    // ---- causal mask + scale + online softmax ----
    float pA[8], pB[8];
#pragma unroll
    for (int r = 0; r < 8; ++r) {
      const int qi = qbase + hoff + r;
      float vA = (kv0 + nloc      <= qi) ? sa[r] * scale : -1e30f;
      float vB = (kv0 + 16 + nloc <= qi) ? sb[r] * scale : -1e30f;
      float rm = fmaxf(vA, vB);
      for (int off = 8; off; off >>= 1) rm = fmaxf(rm, __shfl_xor(rm, off, 32));
      const float mnew  = fmaxf(mrow[r], rm);
      const float alpha = __expf(mrow[r] - mnew);
      vA = __expf(vA - mnew);
      vB = __expf(vB - mnew);
      float rs = vA + vB;
      for (int off = 8; off; off >>= 1) rs += __shfl_xor(rs, off, 32);
      lrow[r] = lrow[r] * alpha + rs;
      mrow[r] = mnew;
      c0[r] *= alpha; c1[r] *= alpha; c2[r] *= alpha; c3[r] *= alpha;
      pA[r] = vA; pB[r] = vB;
    }
    // ---- P -> LDS (bf16, row stride 40) then reload as A-fragment ----
#pragma unroll
    for (int r = 0; r < 8; ++r) {
      pl[(hoff + r) * 40 + nloc]      = f2bf(pA[r]);
      pl[(hoff + r) * 40 + 16 + nloc] = f2bf(pB[r]);
    }
    const v16bf pf = load_a_frag(pl, 40, lane);            // wave-local LDS: in-order DS
    // ---- ctx += P @ V  (V transposed: contiguous along kv) ----
    c0 = WMMA_BF16(pf, load_b_frag(vrow + kv0,                     SEQ, lane), c0);
    c1 = WMMA_BF16(pf, load_b_frag(vrow + (size_t)16 * SEQ + kv0,  SEQ, lane), c1);
    c2 = WMMA_BF16(pf, load_b_frag(vrow + (size_t)32 * SEQ + kv0,  SEQ, lane), c2);
    c3 = WMMA_BF16(pf, load_b_frag(vrow + (size_t)48 * SEQ + kv0,  SEQ, lane), c3);
  }

  // ---- normalize and write ctx as bf16 [token, D] ----
  unsigned short* crow = ctxb + (size_t)(tok0 + qbase) * D_MODEL + h * HDIM;
#pragma unroll
  for (int r = 0; r < 8; ++r) {
    const float inv = 1.0f / lrow[r];
    const size_t ro = (size_t)(hoff + r) * D_MODEL;
    crow[ro + nloc]      = f2bf(c0[r] * inv);
    crow[ro + 16 + nloc] = f2bf(c1[r] * inv);
    crow[ro + 32 + nloc] = f2bf(c2[r] * inv);
    crow[ro + 48 + nloc] = f2bf(c3[r] * inv);
  }
}

// ---------------------------------------------------------------------------
// Residual + LayerNorm over D=512 (one 256-thread block per row, 2 elems/thr).
// ---------------------------------------------------------------------------
template <bool WB>
__global__ __launch_bounds__(256) void ln_residual_kernel(
    const float* __restrict__ a, const float* __restrict__ bsrc,
    const float* __restrict__ g, const float* __restrict__ beta,
    float* __restrict__ yf, unsigned short* __restrict__ yb) {
  __shared__ float red[256];
  const int tid = threadIdx.x;
  const size_t base = (size_t)blockIdx.x * D_MODEL;
  const float v0 = a[base + tid] + bsrc[base + tid];
  const float v1 = a[base + tid + 256] + bsrc[base + tid + 256];
  red[tid] = v0 + v1;
  __syncthreads();
  for (int off = 128; off > 0; off >>= 1) { if (tid < off) red[tid] += red[tid + off]; __syncthreads(); }
  const float mu = red[0] * (1.0f / D_MODEL);
  __syncthreads();
  const float d0 = v0 - mu, d1 = v1 - mu;
  red[tid] = d0 * d0 + d1 * d1;
  __syncthreads();
  for (int off = 128; off > 0; off >>= 1) { if (tid < off) red[tid] += red[tid + off]; __syncthreads(); }
  const float rs = rsqrtf(red[0] * (1.0f / D_MODEL) + 1e-5f);
  const float y0 = d0 * rs * g[tid] + beta[tid];
  const float y1 = d1 * rs * g[tid + 256] + beta[tid + 256];
  yf[base + tid] = y0;
  yf[base + tid + 256] = y1;
  if (WB) { yb[base + tid] = f2bf(y0); yb[base + tid + 256] = f2bf(y1); }
}

// ---------------------------------------------------------------------------
extern "C" void kernel_launch(void* const* d_in, const int* in_sizes, int n_in,
                              void* d_out, int out_size, void* d_ws, size_t ws_size,
                              hipStream_t stream) {
  const float* x    = (const float*)d_in[0];
  const float* Wqkv = (const float*)d_in[1];
  const float* bqkv = (const float*)d_in[2];
  const float* Wo   = (const float*)d_in[3];
  const float* bo   = (const float*)d_in[4];
  const float* W1   = (const float*)d_in[5];
  const float* b1   = (const float*)d_in[6];
  const float* W2   = (const float*)d_in[7];
  const float* b2   = (const float*)d_in[8];
  const float* g1   = (const float*)d_in[9];
  const float* bn1  = (const float*)d_in[10];
  const float* g2   = (const float*)d_in[11];
  const float* bn2  = (const float*)d_in[12];
  (void)in_sizes; (void)n_in; (void)out_size; (void)ws_size;

  char* p = (char*)d_ws;
  auto carve = [&](size_t bytes) { char* r = p; p += (bytes + 255) & ~(size_t)255; return r; };

  unsigned short* xb    = (unsigned short*)carve((size_t)NTOK * D_MODEL * 2);
  unsigned short* wqkvb = (unsigned short*)carve((size_t)QKV_LD * D_MODEL * 2);
  unsigned short* wob   = (unsigned short*)carve((size_t)D_MODEL * D_MODEL * 2);
  unsigned short* w1b   = (unsigned short*)carve((size_t)FDIM * D_MODEL * 2);
  unsigned short* w2b   = (unsigned short*)carve((size_t)D_MODEL * FDIM * 2);
  unsigned short* qkvb  = (unsigned short*)carve((size_t)NTOK * QKV_LD * 2);
  unsigned short* vt    = (unsigned short*)carve((size_t)BATCH * NHEADS * HDIM * SEQ * 2);
  unsigned short* ctxb  = (unsigned short*)carve((size_t)NTOK * D_MODEL * 2);
  float*          attnf = (float*)carve((size_t)NTOK * D_MODEL * 4);
  float*          x1f   = (float*)carve((size_t)NTOK * D_MODEL * 4);
  unsigned short* x1b   = (unsigned short*)carve((size_t)NTOK * D_MODEL * 2);
  unsigned short* ff1b  = (unsigned short*)carve((size_t)NTOK * FDIM * 2);
  float*          ff2f  = (float*)carve((size_t)NTOK * D_MODEL * 4);

  auto castN = [&](const float* s, unsigned short* d, int n) {
    cast_bf16_kernel<<<(n + 2047) / 2048, 256, 0, stream>>>(s, d, n);
  };
  castN(x,    xb,    NTOK * D_MODEL);
  castN(Wqkv, wqkvb, QKV_LD * D_MODEL);
  castN(Wo,   wob,   D_MODEL * D_MODEL);
  castN(W1,   w1b,   FDIM * D_MODEL);
  castN(W2,   w2b,   D_MODEL * FDIM);

  auto gemm_blocks = [](int M, int N) { return (M / GEMM_MB) * (N / GEMM_NB); };

  // 1) fused QKV projection (bf16 out only)
  gemm_wmma_kernel<false, false, true><<<gemm_blocks(NTOK, QKV_LD), 256, 0, stream>>>(
      xb, wqkvb, bqkv, nullptr, qkvb, NTOK, QKV_LD, D_MODEL);

  // 2) transpose V for contiguous P@V B-fragments
  vtrans_kernel<<<(BATCH * NHEADS * HDIM * SEQ) / 256, 256, 0, stream>>>(qkvb, vt);

  // 3) causal flash attention -> ctx (bf16)
  attention_kernel<<<(BATCH * NHEADS * (SEQ / 16)) / ATT_WAVES, 32 * ATT_WAVES, 0, stream>>>(
      qkvb, vt, ctxb);

  // 4) output projection (f32 out)
  gemm_wmma_kernel<false, true, false><<<gemm_blocks(NTOK, D_MODEL), 256, 0, stream>>>(
      ctxb, wob, bo, attnf, nullptr, NTOK, D_MODEL, D_MODEL);

  // 5) residual + LN1 (f32 + bf16 out)
  ln_residual_kernel<true><<<NTOK, 256, 0, stream>>>(x, attnf, g1, bn1, x1f, x1b);

  // 6) FFN up + ReLU (bf16 out)
  gemm_wmma_kernel<true, false, true><<<gemm_blocks(NTOK, FDIM), 256, 0, stream>>>(
      x1b, w1b, b1, nullptr, ff1b, NTOK, FDIM, D_MODEL);

  // 7) FFN down (f32 out)
  gemm_wmma_kernel<false, true, false><<<gemm_blocks(NTOK, D_MODEL), 256, 0, stream>>>(
      ff1b, w2b, b2, ff2f, nullptr, NTOK, D_MODEL, FDIM);

  // 8) residual + LN2 -> d_out (f32)
  ln_residual_kernel<false><<<NTOK, 256, 0, stream>>>(x1f, ff2f, g2, bn2, (float*)d_out, nullptr);
}
